// _DCRNN__56478819942834
// MI455X (gfx1250) — matrix operand order repaired and checked by reading, not compiled, not run on hardware
//
#include <hip/hip_runtime.h>

typedef _Float16 v16h __attribute__((ext_vector_type(16)));
typedef float    v8f  __attribute__((ext_vector_type(8)));
typedef float    v4f  __attribute__((ext_vector_type(4)));

#define FIN  128
#define FOUT 32
// W_z / W_h layout: (2,1,160,32) -> element (d,k,n) at d*5120 + k*32 + n

// Build WMMA B-matrix fragments (hi/lo f16 split of the folded weights),
// pre-swizzled into the 32x16 f16 B lane layout:
//   lane L: N = L&15, K = (L>>4)*8 + i + (i>=8 ? 8 : 0)  (i = 0..15 packed halves)
// frag index = ktile*4 + ntile; 4 ktiles (K=128/32) x 4 ntiles (N=64/16)
__global__ void dcrnn_prep_weights(const float* __restrict__ Wz,
                                   const float* __restrict__ Wh,
                                   _Float16* __restrict__ Bhi,
                                   _Float16* __restrict__ Blo) {
    int t = blockIdx.x * blockDim.x + threadIdx.x;
    if (t >= 512) return;                 // 4*4 frags * 32 lanes
    int lane  = t & 31;
    int ntile = (t >> 5) & 3;
    int ktile = t >> 7;
    int j = lane & 15;
    int g = lane >> 4;
    int N = ntile * 16 + j;               // 0..63 ; <32 -> Z gate, >=32 -> H~
    int kbase = ktile * 32 + g * 8;
    const float* W = (N < FOUT) ? Wz : Wh;
    int col = (N < FOUT) ? N : (N - FOUT);
    int fragbase = ((ktile * 4 + ntile) * 32 + lane) * 16;
#pragma unroll
    for (int i = 0; i < 16; ++i) {
        int K = kbase + i + (i >= 8 ? 8 : 0);                // K < 128 (H0 part is zero)
        float w = W[K * FOUT + col] + W[5120 + K * FOUT + col]; // fold both diffusion dirs
        _Float16 hi = (_Float16)w;
        _Float16 lo = (_Float16)(w - (float)hi);
        Bhi[fragbase + i] = hi;
        Blo[fragbase + i] = lo;
    }
}

// One wave32 per 16-row tile: C[16,64] = X[16,128] @ Wc[128,64] via
// v_wmma_f32_16x16x32_f16 with 2-term hi/lo split, fused GRU epilogue,
// cross-lane reduction for the final [16,1] output.
__global__ void __launch_bounds__(128)
dcrnn_fused(const float* __restrict__ x,
            const _Float16* __restrict__ Bhi,
            const _Float16* __restrict__ Blo,
            const float* __restrict__ bz,
            const float* __restrict__ bh,
            const float* __restrict__ wlin,
            const float* __restrict__ blin,
            float* __restrict__ out,
            int nrows) {
    int wave = threadIdx.x >> 5;
    int lane = threadIdx.x & 31;
    int tile = blockIdx.x * 4 + wave;
    int row0 = tile * 16;
    if (row0 >= nrows) return;            // whole-wave uniform: EXEC stays all-1s

    int j = lane & 15;                    // N index (C layout) / A row (A layout)
    int g = lane >> 4;                    // half-wave group

    // Bias directly in the accumulators: N is constant across a lane's 8 C VGPRs.
    v8f acc[4];
    {
        float bz0 = bz[j], bz1 = bz[j + 16];
        float bh0 = bh[j], bh1 = bh[j + 16];
#pragma unroll
        for (int v = 0; v < 8; ++v) {
            acc[0][v] = bz0; acc[1][v] = bz1;
            acc[2][v] = bh0; acc[3][v] = bh1;
        }
    }

    int row = row0 + j;
    if (row >= nrows) row = nrows - 1;    // safe duplicate read (never hit: 200000%16==0)
    const float* xrow = x + (size_t)row * FIN + g * 8;

#pragma unroll
    for (int kt = 0; kt < 4; ++kt) {
        // A fragment, 16x32 f16 layout: lanes 0-15 K = {0..7,16..23}, lanes 16-31 +8
        v4f lda[4];
        lda[0] = __builtin_nontemporal_load((const v4f*)(xrow + kt * 32 + 0));
        lda[1] = __builtin_nontemporal_load((const v4f*)(xrow + kt * 32 + 4));
        lda[2] = __builtin_nontemporal_load((const v4f*)(xrow + kt * 32 + 16));
        lda[3] = __builtin_nontemporal_load((const v4f*)(xrow + kt * 32 + 20));
        v16h ahi, alo;
#pragma unroll
        for (int q = 0; q < 4; ++q)
#pragma unroll
            for (int e = 0; e < 4; ++e) {
                float f = lda[q][e];
                _Float16 hi = (_Float16)f;
                ahi[q * 4 + e] = hi;
                alo[q * 4 + e] = (_Float16)(f - (float)hi);
            }
#pragma unroll
        for (int nt = 0; nt < 4; ++nt) {
            const v16h bhf = *(const v16h*)(Bhi + ((kt * 4 + nt) * 32 + lane) * 16);
            const v16h blf = *(const v16h*)(Blo + ((kt * 4 + nt) * 32 + lane) * 16);
            acc[nt] = __builtin_amdgcn_wmma_f32_16x16x32_f16(
                false, ahi, false, bhf, (short)0, acc[nt], false, false);
            acc[nt] = __builtin_amdgcn_wmma_f32_16x16x32_f16(
                false, alo, false, bhf, (short)0, acc[nt], false, false);
            acc[nt] = __builtin_amdgcn_wmma_f32_16x16x32_f16(
                false, ahi, false, blf, (short)0, acc[nt], false, false);
        }
    }

    // Epilogue: z = sigmoid(acc01), ht = tanh(acc23), h = relu((1-z)*ht),
    // out[m] = sum_j h_j * wlin_j + blin. (m, j) and (m, j+32) share lane/VGPR.
    float wl0 = wlin[j], wl1 = wlin[j + 16];
    float bl = blin[0];
#pragma unroll
    for (int v = 0; v < 8; ++v) {
        float z0 = 1.0f / (1.0f + expf(-acc[0][v]));
        float z1 = 1.0f / (1.0f + expf(-acc[1][v]));
        float t0 = tanhf(acc[2][v]);
        float t1 = tanhf(acc[3][v]);
        float h0 = fmaxf((1.0f - z0) * t0, 0.0f);
        float h1 = fmaxf((1.0f - z1) * t1, 0.0f);
        float s = h0 * wl0 + h1 * wl1;
        // reduce over j=0..15 within each 16-lane half (xor of bits 0-3)
#pragma unroll
        for (int m = 1; m < 16; m <<= 1) s += __shfl_xor(s, m, 32);
        int mrow = row0 + g * 8 + v;      // C layout: M = v + 8*g
        if (j == 0 && mrow < nrows) out[mrow] = s + bl;
    }
}

extern "C" void kernel_launch(void* const* d_in, const int* in_sizes, int n_in,
                              void* d_out, int out_size, void* d_ws, size_t ws_size,
                              hipStream_t stream) {
    const float* x    = (const float*)d_in[0];
    // d_in[1] edge_index, d_in[2] edge_weight: dead (K=1 -> no propagation)
    const float* Wz   = (const float*)d_in[3];
    const float* bz   = (const float*)d_in[4];
    // d_in[5] W_r, d_in[6] b_r: dead (R only multiplies H0 == 0)
    const float* Wh   = (const float*)d_in[7];
    const float* bh   = (const float*)d_in[8];
    const float* wlin = (const float*)d_in[9];
    const float* blin = (const float*)d_in[10];
    float* out = (float*)d_out;

    _Float16* Bhi = (_Float16*)d_ws;      // 4*4 frags * 32 lanes * 16 halves = 16KB
    _Float16* Blo = Bhi + 4 * 4 * 32 * 16;

    dcrnn_prep_weights<<<2, 256, 0, stream>>>(Wz, Wh, Bhi, Blo);

    int nrows  = in_sizes[0] / FIN;       // 200000
    int tiles  = (nrows + 15) / 16;       // 12500
    int blocks = (tiles + 3) / 4;         // 4 waves (tiles) per 128-thread block
    dcrnn_fused<<<blocks, 128, 0, stream>>>(x, Bhi, Blo, bz, bh, wlin, blin, out, nrows);
}
